// CrossAttention_72310069395911
// MI455X (gfx1250) — matrix operand order repaired and checked by reading, not compiled, not run on hardware
//
#include <hip/hip_runtime.h>

// ---------------------------------------------------------------------------
// CrossAttention forward for MI455X (gfx1250): wave32, bf16 WMMA, TDM staging.
// ---------------------------------------------------------------------------

typedef __attribute__((ext_vector_type(16))) __bf16 v16bf;
typedef __attribute__((ext_vector_type(8)))  float  v8f;
typedef int v4i __attribute__((ext_vector_type(4)));
typedef int v8i __attribute__((ext_vector_type(8)));

#define D_MODEL 1024
#define N_HEADS 16
#define D_HEAD  64
#define SEQ     1024
#define BATCH   4
#define SCALE_F 0.125f      // 1/sqrt(64)
#define EPS_SM  0.02f
#define SSTRIDE 1032        // padded f32 stride for the S/P row block in LDS

// ---- bf16 <-> f32 helpers (round-to-nearest-even) -------------------------
__device__ __forceinline__ unsigned short f2bf(float f) {
  union { float f; unsigned u; } x; x.f = f;
  unsigned r = x.u + 0x7FFFu + ((x.u >> 16) & 1u);
  return (unsigned short)(r >> 16);
}
__device__ __forceinline__ float bf2f(unsigned short s) {
  union { float f; unsigned u; } x; x.u = ((unsigned)s) << 16; return x.f;
}

union FragU { v16bf v; unsigned u[8]; };

// ---- WMMA fragment loaders per CDNA5 ISA 7.12.2 layouts --------------------
// A (16x32 bf16, MxK): lane holds row M=lane&15; K-half select by lane>>4.
// Compiles to two contiguous ds_load_b128 per fragment.
__device__ __forceinline__ v16bf load_frag_a_bf(const unsigned short* lds,
                                                int row0, int k0, int ld) {
  const int lane = threadIdx.x & 31;
  const int row  = row0 + (lane & 15);
  const int kh   = (lane >> 4) << 3;
  const unsigned* base = (const unsigned*)(lds + row * ld + k0);
  FragU f;
#pragma unroll
  for (int j = 0; j < 8; ++j) {
    const int kk = (j < 4) ? (2 * j + kh) : (16 + 2 * (j - 4) + kh);
    f.u[j] = base[kk >> 1];
  }
  return f.v;
}

// B (32x16 bf16, KxN) from LDS stored [n][k] row-major: lane holds col N;
// lanes 0-15 K=0..7,16..23, lanes 16-31 K=8..15,24..31 (matches A).
__device__ __forceinline__ v16bf load_frag_b_bf(const unsigned short* lds,
                                                int col0, int k0, int ld) {
  const int lane = threadIdx.x & 31;
  const int col  = col0 + (lane & 15);
  const int kb   = (lane >> 4) << 2;   // dword offset: 0 or 4
  const unsigned* base = (const unsigned*)(lds + col * ld + k0);
  FragU f;
#pragma unroll
  for (int j = 0; j < 8; ++j) f.u[j] = base[(j < 4) ? (kb + j) : (8 + kb + (j - 4))];
  return f.v;
}

// A fragment built from an f32 LDS matrix (P matrix in attention), cvt->bf16.
__device__ __forceinline__ v16bf load_frag_a_f32(const float* S,
                                                 int row0, int k0, int ld) {
  const int lane = threadIdx.x & 31;
  const int row  = row0 + (lane & 15);
  const int kh   = (lane >> 4) << 3;
  const float* base = S + row * ld + k0;
  FragU f;
#pragma unroll
  for (int j = 0; j < 8; ++j) {
    const int kk = (j < 4) ? (2 * j + kh) : (16 + 2 * (j - 4) + kh);
    f.u[j] = (unsigned)f2bf(base[kk]) | ((unsigned)f2bf(base[kk + 1]) << 16);
  }
  return f.v;
}

__device__ __forceinline__ v8f wmma_bf16(v16bf a, v16bf b, v8f c) {
  return __builtin_amdgcn_wmma_f32_16x16x32_bf16(
      /*neg_a=*/false, a, /*neg_b=*/false, b,
      /*c_mod=*/(short)0, c, /*reuse_a=*/false, /*reuse_b=*/false);
}

// ---------------------------------------------------------------------------
// Tensor Data Mover: 1-D contiguous bf16 tile, global -> LDS.
// D# per CDNA5 ISA ch.8: group0 = {count=1 | lds_addr | global_addr | type=2},
// group1 = {data_size=2B, tensor_dim0=tile_dim0=nelems, dims1..=1/0}.
// Issued wave-uniform (descriptor in SGPRs); tracked by TENSORcnt.
// ---------------------------------------------------------------------------
__device__ __forceinline__ void tdm_load_to_lds_1d(void* lds_dst,
                                                   const void* gsrc,
                                                   unsigned nelems) {
  const unsigned lds_off = (unsigned)(uintptr_t)lds_dst;   // addr[31:0] = LDS byte addr
  const unsigned long long ga = (unsigned long long)(uintptr_t)gsrc;
  v4i g0;
  g0[0] = 1;                                        // count=1 (valid user D#)
  g0[1] = (int)lds_off;                             // lds_addr
  g0[2] = (int)(unsigned)(ga & 0xFFFFFFFFull);      // global_addr[31:0]
  g0[3] = (int)((unsigned)((ga >> 32) & 0x1FFFFFFull) | 0x80000000u); // [56:32]|type=2
  v8i g1;
  g1[0] = 0x00010000;                               // data_size=1 (2 bytes)
  g1[1] = (int)((nelems & 0xFFFFu) << 16);          // tensor_dim0[15:0] @ bits63:48
  g1[2] = (int)((nelems >> 16) & 0xFFFFu) | (1 << 16); // tensor_dim0 hi | tensor_dim1=1
  g1[3] = (int)((nelems & 0xFFFFu) << 16);          // tile_dim0 = nelems
  g1[4] = 1;                                        // tile_dim1 = 1
  g1[5] = (int)nelems;                              // tensor_dim0_stride
  g1[6] = 0;
  g1[7] = 0;
  const v4i gz = {0, 0, 0, 0};
  asm volatile("tensor_load_to_lds %0, %1, %2, %3"
               :: "s"(g0), "s"(g1), "s"(gz), "s"(gz)
               : "memory");
}

// ---------------------------------------------------------------------------
// Kernel 1: C[M=4096, N=1024] = A @ W^T + bias.
//   mode 0: bf16 out, [B,H,L,64]          (Q, K)
//   mode 2: bf16 out, [B,H,kblk,d,key]    (V, pre-transposed 64x64 tiles)
//   mode 1: f32 out, row-major [M,N]      (final projection)
// 128x64 block tile, 8 waves = 4(M) x 2(N), each wave 32x32 (4 WMMA acc).
// ---------------------------------------------------------------------------
__global__ __launch_bounds__(256) void proj_gemm_kernel(
    const float* __restrict__ A, const float* __restrict__ W,
    const float* __restrict__ bias, unsigned short* __restrict__ out_bf,
    float* __restrict__ out_row, int mode) {
  __shared__ unsigned short sA[128 * 32];   // 8 KB
  __shared__ unsigned short sB[64 * 32];    // 4 KB

  const int tid  = threadIdx.x;
  const int lane = tid & 31;
  const int wave = tid >> 5;
  const int m0 = blockIdx.x * 128;
  const int n0 = blockIdx.y * 64;
  const int mr = (wave & 3) << 5;       // wave M offset (32 rows)
  const int nh = (wave >> 2) << 5;      // wave N offset (32 cols)

  v8f acc[2][2] = {};

  for (int kk = 0; kk < D_MODEL; kk += 32) {
#pragma unroll
    for (int j = 0; j < 16; ++j) {      // 4096 A elems / 256 threads
      const int e = tid + 256 * j;
      const int r = e >> 5, c = e & 31;
      sA[r * 32 + c] = f2bf(A[(size_t)(m0 + r) * D_MODEL + kk + c]);
    }
#pragma unroll
    for (int j = 0; j < 8; ++j) {       // 2048 W elems / 256 threads
      const int e = tid + 256 * j;
      const int r = e >> 5, c = e & 31;
      sB[r * 32 + c] = f2bf(W[(size_t)(n0 + r) * D_MODEL + kk + c]);
    }
    __syncthreads();
    v16bf a0 = load_frag_a_bf(sA, mr, 0, 32);
    v16bf a1 = load_frag_a_bf(sA, mr + 16, 0, 32);
    v16bf b0 = load_frag_b_bf(sB, nh, 0, 32);
    v16bf b1 = load_frag_b_bf(sB, nh + 16, 0, 32);
    acc[0][0] = wmma_bf16(a0, b0, acc[0][0]);
    acc[0][1] = wmma_bf16(a0, b1, acc[0][1]);
    acc[1][0] = wmma_bf16(a1, b0, acc[1][0]);
    acc[1][1] = wmma_bf16(a1, b1, acc[1][1]);
    __syncthreads();
  }

#pragma unroll
  for (int i = 0; i < 2; ++i) {
#pragma unroll
    for (int t = 0; t < 2; ++t) {
#pragma unroll
      for (int r = 0; r < 8; ++r) {
        const int m = m0 + mr + i * 16 + ((lane >> 4) << 3) + r;
        const int n = n0 + nh + t * 16 + (lane & 15);
        const float v = acc[i][t][r] + bias[n];
        if (mode == 1) {
          out_row[(size_t)m * D_MODEL + n] = v;
        } else {
          const int bb = m >> 10, ll = m & 1023;
          const int hh = n >> 6,  dd = n & 63;
          if (mode == 0) {
            out_bf[(size_t)((bb * N_HEADS + hh) * SEQ + ll) * D_HEAD + dd] =
                f2bf(v);
          } else {  // mode 2: V in [B,H, key-block, d, key-in-block] tiles
            out_bf[(((size_t)(bb * N_HEADS + hh) * 16 + (ll >> 6)) * 64 + dd) *
                       64 + (ll & 63)] = f2bf(v);
          }
        }
      }
    }
  }
}

// ---------------------------------------------------------------------------
// Kernel 2: per-head-vector L2 normalize of Q and K (bf16 in place).
// One wave32 per 64-element row; each lane owns one packed dword (2 bf16).
// ---------------------------------------------------------------------------
__global__ __launch_bounds__(256) void l2norm_kernel(
    unsigned short* __restrict__ q, unsigned short* __restrict__ k) {
  const int wave = threadIdx.x >> 5, lane = threadIdx.x & 31;
  const int row = blockIdx.x * 8 + wave;                // 0 .. 2*B*H*L-1
  const int nQ  = BATCH * N_HEADS * SEQ;                // 65536
  unsigned short* p =
      (row < nQ) ? (q + (size_t)row * D_HEAD) : (k + (size_t)(row - nQ) * D_HEAD);
  unsigned* p32 = (unsigned*)p;
  unsigned w = p32[lane];
  float x0 = bf2f((unsigned short)(w & 0xFFFFu));
  float x1 = bf2f((unsigned short)(w >> 16));
  float ss = x0 * x0 + x1 * x1;
#pragma unroll
  for (int i = 16; i > 0; i >>= 1) ss += __shfl_xor(ss, i, 32);
  const float sc = 1.0f / fmaxf(sqrtf(ss), 1e-12f);
  x0 *= sc; x1 *= sc;
  p32[lane] = (unsigned)f2bf(x0) | ((unsigned)f2bf(x1) << 16);
}

// ---------------------------------------------------------------------------
// Kernel 3: attention for one (b, h, 64-query tile).  ~286KB dynamic LDS
// (CDNA5: up to 320KB per workgroup) holds the full 64x1024 S/P row block.
// Q/K/V 8KB tiles are staged with the Tensor Data Mover (wave 0 issues,
// s_wait_tensorcnt, workgroup barrier releases consumers).
// ---------------------------------------------------------------------------
__global__ __launch_bounds__(256) void attn_kernel(
    const unsigned short* __restrict__ qb, const unsigned short* __restrict__ kb,
    const unsigned short* __restrict__ vt, const int* __restrict__ mask,
    float* __restrict__ ctx) {
  extern __shared__ char smem[];
  float*          sS = (float*)smem;                               // 64*SSTRIDE f32
  unsigned short* sQ = (unsigned short*)(smem + 64 * SSTRIDE * 4); // 64*64 bf16
  unsigned short* sK = sQ + 64 * 64;                               // 64*64 bf16
  unsigned short* sV = sK + 64 * 64;                               // [d][key] bf16
  float*       sValid = (float*)(sV + 64 * 64);                    // 1024 f32
  float*       sStat  = sValid + SEQ;                              // invN

  const int tid = threadIdx.x, lane = tid & 31, wave = tid >> 5;
  const int bh = blockIdx.x >> 4;       // (b*H + h)
  const int q0 = (blockIdx.x & 15) * 64;
  const int b = bh >> 4;
  const int h = bh & 15;
  const size_t headBase = (size_t)bh * SEQ * D_HEAD;

  // -- stage mask validity (VALU path) + Q tile (TDM) ------------------------
#pragma unroll
  for (int j = 0; j < 4; ++j) {
    const int c = tid + 256 * j;
    sValid[c] = (mask[b * SEQ + c] != 0) ? 0.0f : 1.0f;
  }
  if (wave == 0) {
    tdm_load_to_lds_1d(sQ, qb + headBase + (size_t)q0 * D_HEAD, 64 * 64);
    __builtin_amdgcn_s_wait_tensorcnt(0);
  }
  __syncthreads();
  if (wave == 0) {                      // count valid keys for smoothing term
    float s = 0.0f;
#pragma unroll
    for (int i = 0; i < 32; ++i) s += sValid[lane + 32 * i];
#pragma unroll
    for (int i = 16; i > 0; i >>= 1) s += __shfl_xor(s, i, 32);
    if (lane == 0) sStat[0] = 1.0f / fmaxf(s, 1.0f);
  }
  __syncthreads();

  const int mr = (wave & 3) << 4;       // wave's query sub-tile
  const int nh = (wave >> 2) << 5;      // wave's column half
  const v16bf aq0 = load_frag_a_bf(sQ, mr, 0, 64);
  const v16bf aq1 = load_frag_a_bf(sQ, mr, 32, 64);

  // -- Phase 1: masked, scaled S = Q K^T -------------------------------------
  for (int kblk = 0; kblk < 16; ++kblk) {
    if (wave == 0) {
      tdm_load_to_lds_1d(sK, kb + headBase + (size_t)kblk * 64 * D_HEAD, 64 * 64);
      __builtin_amdgcn_s_wait_tensorcnt(0);
    }
    __syncthreads();
#pragma unroll
    for (int jt = 0; jt < 2; ++jt) {
      const int ncol = nh + jt * 16;
      v16bf b0 = load_frag_b_bf(sK, ncol, 0, 64);
      v16bf b1 = load_frag_b_bf(sK, ncol, 32, 64);
      v8f s = {};
      s = wmma_bf16(aq0, b0, s);
      s = wmma_bf16(aq1, b1, s);
#pragma unroll
      for (int r = 0; r < 8; ++r) {
        const int m = mr + ((lane >> 4) << 3) + r;
        const int n = kblk * 64 + ncol + (lane & 15);
        float v = s[r] * SCALE_F;
        if (sValid[n] == 0.0f) v = -1.0e9f;
        sS[m * SSTRIDE + n] = v;
      }
    }
    __syncthreads();
  }

  // -- Phase 2: row softmax + pad renorm + smoothing (in place, f32) ---------
  const float invN = sStat[0];
  for (int rr = 0; rr < 8; ++rr) {
    float* Srow = sS + (wave * 8 + rr) * SSTRIDE;
    float mx = -3.0e38f;
    for (int i = 0; i < 32; ++i) mx = fmaxf(mx, Srow[lane + 32 * i]);
#pragma unroll
    for (int i = 16; i > 0; i >>= 1) mx = fmaxf(mx, __shfl_xor(mx, i, 32));
    float se = 0.0f;
    for (int i = 0; i < 32; ++i) {
      const int c = lane + 32 * i;
      const float e = __expf(Srow[c] - mx) * sValid[c];
      Srow[c] = e;
      se += e;
    }
#pragma unroll
    for (int i = 16; i > 0; i >>= 1) se += __shfl_xor(se, i, 32);
    const float inv = 1.0f / fmaxf(se, 1e-8f);
    for (int i = 0; i < 32; ++i) {
      const int c = lane + 32 * i;
      Srow[c] = (1.0f - EPS_SM) * (Srow[c] * inv) + EPS_SM * sValid[c] * invN;
    }
  }
  __syncthreads();

  // -- Phase 3: O = P V  (V tiles already [d][key]-transposed in memory) -----
  v8f o0 = {};
  v8f o1 = {};
  for (int kblk = 0; kblk < 16; ++kblk) {
    if (wave == 0) {
      tdm_load_to_lds_1d(sV, vt + ((size_t)bh * 16 + kblk) * 64 * 64, 64 * 64);
      __builtin_amdgcn_s_wait_tensorcnt(0);
    }
    __syncthreads();
    const v16bf ap0 = load_frag_a_f32(sS, mr, kblk * 64, SSTRIDE);
    const v16bf ap1 = load_frag_a_f32(sS, mr, kblk * 64 + 32, SSTRIDE);
    o0 = wmma_bf16(ap0, load_frag_b_bf(sV, nh, 0, 64), o0);
    o0 = wmma_bf16(ap1, load_frag_b_bf(sV, nh, 32, 64), o0);
    o1 = wmma_bf16(ap0, load_frag_b_bf(sV, nh + 16, 0, 64), o1);
    o1 = wmma_bf16(ap1, load_frag_b_bf(sV, nh + 16, 32, 64), o1);
    __syncthreads();
  }

  // -- store context directly in [B, L, D_MODEL] -----------------------------
#pragma unroll
  for (int r = 0; r < 8; ++r) {
    const int m = mr + ((lane >> 4) << 3) + r;
    const size_t rowOff = ((size_t)b * SEQ + q0 + m) * D_MODEL + h * D_HEAD;
    ctx[rowOff + nh + (lane & 15)]      = o0[r];
    ctx[rowOff + nh + 16 + (lane & 15)] = o1[r];
  }
}

// ---------------------------------------------------------------------------
// Host launch
// ---------------------------------------------------------------------------
extern "C" void kernel_launch(void* const* d_in, const int* in_sizes, int n_in,
                              void* d_out, int out_size, void* d_ws,
                              size_t ws_size, hipStream_t stream) {
  (void)in_sizes; (void)n_in; (void)out_size; (void)ws_size;
  const float* q_in  = (const float*)d_in[0];
  const float* kv_in = (const float*)d_in[1];
  const int*   mask  = (const int*)d_in[2];
  const float* Wq = (const float*)d_in[3];
  const float* bq = (const float*)d_in[4];
  const float* Wk = (const float*)d_in[5];
  const float* bk = (const float*)d_in[6];
  const float* Wv = (const float*)d_in[7];
  const float* bv = (const float*)d_in[8];
  const float* Wo = (const float*)d_in[9];
  const float* bo = (const float*)d_in[10];
  float* out = (float*)d_out;

  // workspace: q/k bf16 [B,H,L,64], v bf16 [B,H,16,64,64] (8 MiB each),
  //            ctx f32 [B,L,1024] (16 MiB)
  char* ws = (char*)d_ws;
  unsigned short* qbf = (unsigned short*)(ws);
  unsigned short* kbf = (unsigned short*)(ws + ((size_t)8 << 20));
  unsigned short* vbf = (unsigned short*)(ws + ((size_t)16 << 20));
  float*          ctx = (float*)(ws + ((size_t)24 << 20));

  const dim3 blk(256);
  const dim3 gG(32, 16);   // M=4096 / 128, N=1024 / 64

  proj_gemm_kernel<<<gG, blk, 0, stream>>>(q_in,  Wq, bq, qbf, nullptr, 0);
  proj_gemm_kernel<<<gG, blk, 0, stream>>>(kv_in, Wk, bk, kbf, nullptr, 0);
  proj_gemm_kernel<<<gG, blk, 0, stream>>>(kv_in, Wv, bv, vbf, nullptr, 2);

  l2norm_kernel<<<dim3((2 * BATCH * N_HEADS * SEQ) / 8), blk, 0, stream>>>(qbf, kbf);

  const size_t smem = (size_t)64 * SSTRIDE * 4   // S/P block
                    + (size_t)3 * 64 * 64 * 2    // Q, K, V tiles
                    + (size_t)SEQ * 4 + 64;      // valid mask + stats
  (void)hipFuncSetAttribute(reinterpret_cast<const void*>(attn_kernel),
                            hipFuncAttributeMaxDynamicSharedMemorySize,
                            (int)smem);
  attn_kernel<<<dim3(BATCH * N_HEADS * (SEQ / 64)), blk, smem, stream>>>(
      qbf, kbf, vbf, mask, ctx);

  proj_gemm_kernel<<<gG, blk, 0, stream>>>(ctx, Wo, bo, nullptr, out, 1);
}
